// MemTransformerLM_50594714747099
// MI455X (gfx1250) — matrix operand order repaired and checked by reading, not compiled, not run on hardware
//
#include <hip/hip_runtime.h>

#define QLEN   1024
#define KLEN   2048
#define BATCH  4
#define NHEAD  16
#define DHEAD  64
#define MEML   1024
#define DMODEL 1024
#define DINNER 4096

typedef __attribute__((ext_vector_type(16))) __bf16 v16bf;
typedef __attribute__((ext_vector_type(8)))  float  v8f;
typedef __attribute__((ext_vector_type(4)))  int    v4i;

#define AS1 __attribute__((address_space(1)))
#define AS3 __attribute__((address_space(3)))

// ---------- CDNA5 async global->LDS copy (guarded; falls back to sync) ------

#if __has_builtin(__builtin_amdgcn_global_load_async_to_lds_b128)
#define HAVE_ASYNC_LDS 1
__device__ __forceinline__ void async_copy_b128(unsigned* lds, const unsigned* g) {
  __builtin_amdgcn_global_load_async_to_lds_b128((AS1 v4i*)g, (AS3 v4i*)lds, 0, 0);
}
#else
#define HAVE_ASYNC_LDS 0
__device__ __forceinline__ void async_copy_b128(unsigned* lds, const unsigned* g) {
  *(uint4*)lds = *(const uint4*)g;
}
#endif

__device__ __forceinline__ void async_wait_all() {
#if HAVE_ASYNC_LDS
#if __has_builtin(__builtin_amdgcn_s_wait_asynccnt)
  __builtin_amdgcn_s_wait_asynccnt(0);
#else
  asm volatile("s_wait_asynccnt 0x0" ::: "memory");
#endif
#endif
}

// ---------- helpers ----------

__device__ __forceinline__ unsigned short f2bf(float x) {
  union { float f; unsigned u; } v; v.f = x;
  unsigned r = v.u + 0x7FFFu + ((v.u >> 16) & 1u);   // RNE
  return (unsigned short)(r >> 16);
}

// A-matrix 16x32 bf16 fragment (contiguous k order; B must match): lane m=lane&15, hi=lane>>4.
// VGPR v holds K=2v..2v+1 (+16*hi).
__device__ __forceinline__ v16bf load_frag_a(const unsigned* p, int ld, int lane) {
  union { unsigned u[8]; v16bf v; } f;
  int m = lane & 15, hi = lane >> 4;
  const unsigned* r = p + (size_t)m * ld + 8 * hi;
#pragma unroll
  for (int i = 0; i < 8; ++i) f.u[i] = r[i];
  return f.v;
}

// B-matrix 32x16 bf16 fragment: lane n=lane&15, hi=lane>>4, VGPR v holds
// K=2v..2v+1 (+16*hi).  Source must have K-pairs adjacent ("NT" storage
// [n][K] or the swizzle-packed LDS layout).
__device__ __forceinline__ v16bf load_frag_b(const unsigned* p, int ld, int lane) {
  union { unsigned u[8]; v16bf v; } f;
  int n = lane & 15, hi = lane >> 4;
  const unsigned* r = p + (size_t)n * ld + 8 * hi;
#pragma unroll
  for (int i = 0; i < 8; ++i) f.u[i] = r[i];
  return f.v;
}

// B-fragment from rel-pos table r[m][d] (NT orientation) with row bounds check.
__device__ __forceinline__ v16bf load_frag_r(const unsigned* base, int mrow, int kuoff, int lane) {
  union { unsigned u[8]; v16bf v; } f;
  int nn = lane & 15, hi = lane >> 4;
  int m = mrow + nn;
  if (m < 0 || m >= KLEN) {
#pragma unroll
    for (int i = 0; i < 8; ++i) f.u[i] = 0u;
  } else {
    const unsigned* r = base + (size_t)m * 32 + kuoff + 8 * hi;
#pragma unroll
    for (int i = 0; i < 8; ++i) f.u[i] = r[i];
  }
  return f.v;
}

__device__ __forceinline__ v8f wmma_bf16(v16bf a, v16bf b, v8f c) {
  return __builtin_amdgcn_wmma_f32_16x16x32_bf16(false, a, false, b, (short)0, c, false, false);
}

// ---------- elementwise kernels ----------

__global__ __launch_bounds__(256) void cast_bf16_kernel(const float* __restrict__ in,
                                                        unsigned short* __restrict__ out, int n) {
  int i = blockIdx.x * 256 + threadIdx.x;
  if (i < n) out[i] = f2bf(in[i]);
}

__global__ __launch_bounds__(256) void bias_relu_cast_kernel(const float* __restrict__ x,
                                                             const float* __restrict__ bias,
                                                             unsigned short* __restrict__ out,
                                                             int ncols) {
  size_t t = (size_t)blockIdx.x * 256 + threadIdx.x;
  int c = (int)(t & (size_t)(ncols - 1));
  float v = x[t] + bias[c];
  out[t] = f2bf(v > 0.f ? v : 0.f);
}

// ---------- generic bf16 GEMM: C[M,N] f32 = A[M,K] @ B[K,N], row-major ------
// Block 128 thr = 4 waves (2x2); workgroup tile 128x64; wave tile 64x32
// (4x2 WMMA tiles -> 8 WMMA per 32-wide k-step).  Double-buffered LDS with
// async global->LDS DMA for the A tile.  M % 128 == 0, N % 64 == 0, K % 32 == 0.

struct BReg { unsigned x[4], y[4]; };

__device__ __forceinline__ void gemm_stage_a(unsigned* aT, const unsigned* A32,
                                             int m0, int ldau, int k0, int tid) {
#pragma unroll
  for (int t = 0; t < 4; ++t) {
    int id = t * 128 + tid;
    int row = id >> 2, c = id & 3;                  // 128 rows x 4 b128 chunks
    async_copy_b128(aT + id * 4,
                    A32 + (size_t)(m0 + row) * ldau + (k0 >> 1) + 4 * c);
  }
}

__device__ __forceinline__ void gemm_load_b(BReg& r, const unsigned* B32,
                                            int n0, int ldbu, int k0, int tid) {
#pragma unroll
  for (int t = 0; t < 4; ++t) {
    int blk = t * 128 + tid; int kp = blk & 15, c = blk >> 4;   // c = n-pair 0..31
    r.x[t] = B32[(size_t)(k0 + 2 * kp) * ldbu + (n0 >> 1) + c];
    r.y[t] = B32[(size_t)(k0 + 2 * kp + 1) * ldbu + (n0 >> 1) + c];
  }
}

__device__ __forceinline__ void gemm_store_b(unsigned* bT, const BReg& r, int tid) {
#pragma unroll
  for (int t = 0; t < 4; ++t) {
    int blk = t * 128 + tid; int kp = blk & 15, c = blk >> 4;
    bT[(2 * c) * 16 + kp]     = (r.x[t] & 0xFFFFu) | (r.y[t] << 16);
    bT[(2 * c + 1) * 16 + kp] = (r.x[t] >> 16)     | (r.y[t] & 0xFFFF0000u);
  }
}

__global__ __launch_bounds__(128) void gemm_bf16_kernel(const unsigned short* __restrict__ A,
                                                        const unsigned short* __restrict__ B,
                                                        float* __restrict__ C,
                                                        int Mdim, int Ndim, int Kdim) {
  __shared__ __align__(16) unsigned aT[2][128 * 16];  // [buf][row][Kpair]
  __shared__ __align__(16) unsigned bT[2][64 * 16];   // [buf][n][Kpair] swizzled
  const unsigned* A32 = (const unsigned*)A;
  const unsigned* B32 = (const unsigned*)B;
  int tid = threadIdx.x, lane = tid & 31, wv = tid >> 5;
  int wr = wv >> 1, wc = wv & 1;
  int m0 = blockIdx.y * 128, n0 = blockIdx.x * 64;
  int ldau = Kdim >> 1, ldbu = Ndim >> 1;
  (void)Mdim;

  v8f zero8 = {0.f, 0.f, 0.f, 0.f, 0.f, 0.f, 0.f, 0.f};
  v8f acc[4][2];
#pragma unroll
  for (int i = 0; i < 4; ++i) { acc[i][0] = zero8; acc[i][1] = zero8; }

  // prologue: stage k-step 0 into buffer 0
  BReg br;
  gemm_stage_a(aT[0], A32, m0, ldau, 0, tid);
  gemm_load_b(br, B32, n0, ldbu, 0, tid);
  async_wait_all();
  gemm_store_b(bT[0], br, tid);

  int cur = 0;
  for (int k0 = 0; k0 < Kdim; k0 += 32) {
    __syncthreads();                                 // cur staging visible
    bool has_next = (k0 + 32) < Kdim;
    if (has_next) {                                  // DMA next tile during math
      gemm_stage_a(aT[cur ^ 1], A32, m0, ldau, k0 + 32, tid);
      gemm_load_b(br, B32, n0, ldbu, k0 + 32, tid);
    }
    v16bf af[4], bfr[2];
#pragma unroll
    for (int i = 0; i < 4; ++i)
      af[i] = load_frag_a(aT[cur] + (wr * 64 + i * 16) * 16, 16, lane);
#pragma unroll
    for (int j = 0; j < 2; ++j)
      bfr[j] = load_frag_b(bT[cur] + (wc * 32 + j * 16) * 16, 16, lane);
#pragma unroll
    for (int i = 0; i < 4; ++i) {
      acc[i][0] = wmma_bf16(af[i], bfr[0], acc[i][0]);
      acc[i][1] = wmma_bf16(af[i], bfr[1], acc[i][1]);
    }
    if (has_next) {
      async_wait_all();
      gemm_store_b(bT[cur ^ 1], br, tid);
    }
    __syncthreads();                                 // cur free for next staging
    cur ^= 1;
  }

  int hi = lane >> 4, nn = lane & 15;
  int mb = m0 + wr * 64, nb = n0 + wc * 32;
#pragma unroll
  for (int i = 0; i < 4; ++i)
#pragma unroll
    for (int j = 0; j < 2; ++j)
#pragma unroll
      for (int v = 0; v < 8; ++v)
        C[(size_t)(mb + i * 16 + v + 8 * hi) * Ndim + nb + j * 16 + nn] = acc[i][j][v];
}

// ---------- repack: w_heads f32 -> rw_q/rr_q (bias folded), K, V in bf16 -----

__global__ __launch_bounds__(256) void repack_qkv_kernel(const float* __restrict__ wheads,
                                                         const float* __restrict__ rwb,
                                                         const float* __restrict__ rrb,
                                                         unsigned short* __restrict__ rwq,
                                                         unsigned short* __restrict__ rrq,
                                                         unsigned short* __restrict__ kb,
                                                         unsigned short* __restrict__ vb) {
  size_t t = (size_t)blockIdx.x * 256 + threadIdx.x;   // < 8192*3072
  int col = (int)(t % 3072);
  int row = (int)(t / 3072);
  int bb = row & (BATCH - 1);
  int jk = row >> 2;
  float val = wheads[t];
  if (col < 1024) {
    if (jk >= MEML) {
      int i = jk - MEML;
      int n = col >> 6, d = col & 63;
      size_t o = (((size_t)(bb * NHEAD + n) * QLEN + i) << 6) + d;
      rwq[o] = f2bf(val + rwb[col]);
      rrq[o] = f2bf(val + rrb[col]);
    }
  } else if (col < 2048) {
    int c = col - 1024; int n = c >> 6, d = c & 63;
    kb[(((size_t)(bb * NHEAD + n) * KLEN + jk) << 6) + d] = f2bf(val);
  } else {
    int c = col - 2048; int n = c >> 6, d = c & 63;
    vb[(((size_t)(bb * NHEAD + n) * KLEN + jk) << 6) + d] = f2bf(val);
  }
}

__global__ __launch_bounds__(256) void repack_r_kernel(const float* __restrict__ rproj,
                                                       unsigned short* __restrict__ rb) {
  size_t t = (size_t)blockIdx.x * 256 + threadIdx.x;   // < 2048*1024
  int m = (int)(t >> 10);
  int col = (int)(t & 1023);
  int n = col >> 6, d = col & 63;
  rb[(((size_t)n * KLEN + m) << 6) + d] = f2bf(rproj[t]);
}

// ---------- flash attention with folded rel_shift ---------------------------
// grid (B*NH, QLEN/64); block 128 = 4 waves; wave owns 16 query rows.
// score[i,j] = scale*( rw_q[i].k[j] + rr_q[i].r[j-i+QLEN-1] ), mask j > i+MEML.

__global__ __launch_bounds__(128) void attn_kernel(const unsigned short* __restrict__ rwq,
                                                   const unsigned short* __restrict__ rrq,
                                                   const unsigned short* __restrict__ kb,
                                                   const unsigned short* __restrict__ vb,
                                                   const unsigned short* __restrict__ rb,
                                                   unsigned short* __restrict__ av) {
  __shared__ __align__(16) unsigned k_lds[32 * 32];   // K tile row-major [j][d-pair]
  __shared__ __align__(16) unsigned v_lds[64 * 16];   // V tile swizzled [d][j-pair]
  __shared__ float    p_lds[4][16 * 48];              // per-wave BD band (16 x 48)
  __shared__ unsigned pr_lds[4][16 * 16];             // per-wave probs 16x32 bf16

  int tid = threadIdx.x, lane = tid & 31, wv = tid >> 5;
  int hi = lane >> 4, ln = lane & 15;
  int bh = blockIdx.x;
  int b = bh >> 4, n = bh & 15;
  int i0 = blockIdx.y * 64 + wv * 16;

  const unsigned* q32rw = (const unsigned*)rwq + (((size_t)bh * QLEN + i0) << 5);
  const unsigned* q32rr = (const unsigned*)rrq + (((size_t)bh * QLEN + i0) << 5);
  const unsigned* k32 = (const unsigned*)kb + ((size_t)bh * KLEN << 5);
  const unsigned* v32 = (const unsigned*)vb + ((size_t)bh * KLEN << 5);
  const unsigned* r32 = (const unsigned*)rb + ((size_t)n * KLEN << 5);

  v16bf arw[2], arr[2];
  arw[0] = load_frag_a(q32rw, 32, lane);
  arw[1] = load_frag_a(q32rw + 16, 32, lane);
  arr[0] = load_frag_a(q32rr, 32, lane);
  arr[1] = load_frag_a(q32rr + 16, 32, lane);

  v8f zero8 = {0.f, 0.f, 0.f, 0.f, 0.f, 0.f, 0.f, 0.f};
  v8f O0 = zero8, O1 = zero8, O2 = zero8, O3 = zero8;
  float mrun[8], lrun[8];
#pragma unroll
  for (int v = 0; v < 8; ++v) { mrun[v] = -1e30f; lrun[v] = 0.f; }

  int jmax = blockIdx.y * 64 + 63 + MEML;             // uniform across block
  if (jmax > KLEN - 1) jmax = KLEN - 1;
  const float scale = 0.125f;                         // 1/sqrt(64)

  for (int j0 = 0; j0 <= jmax; j0 += 32) {
    __syncthreads();
    // K tile: pure copy -> async DMA (2 x b128 per thread)
#pragma unroll
    for (int t = 0; t < 2; ++t) {
      int id = t * 128 + tid;
      int r = id >> 3, c = id & 7;
      async_copy_b128(k_lds + id * 4, k32 + (size_t)(j0 + r) * 32 + 4 * c);
    }
    // V tile: K-pair swizzle, two-phase (all loads, then all stores)
    unsigned vx[4], vy[4];
#pragma unroll
    for (int t = 0; t < 4; ++t) {
      int blk = t * 128 + tid; int kp = blk & 15, c = blk >> 4;
      vx[t] = v32[(size_t)(j0 + 2 * kp) * 32 + c];
      vy[t] = v32[(size_t)(j0 + 2 * kp + 1) * 32 + c];
    }
#pragma unroll
    for (int t = 0; t < 4; ++t) {
      int blk = t * 128 + tid; int kp = blk & 15, c = blk >> 4;
      v_lds[(2 * c) * 16 + kp]     = (vx[t] & 0xFFFFu) | (vy[t] << 16);
      v_lds[(2 * c + 1) * 16 + kp] = (vx[t] >> 16)     | (vy[t] & 0xFFFF0000u);
    }
    async_wait_all();
    __syncthreads();

    // AC: two 16x16 score subtiles over dh=64 (two k=32 chunks each)
    v8f S0 = zero8, S1 = zero8;
#pragma unroll
    for (int c = 0; c < 2; ++c) {
      v16bf bk0 = load_frag_b(k_lds + 0 * 32 + c * 16, 32, lane);
      v16bf bk1 = load_frag_b(k_lds + 16 * 32 + c * 16, 32, lane);
      S0 = wmma_bf16(arw[c], bk0, S0);
      S1 = wmma_bf16(arw[c], bk1, S1);
    }

    // BD band: P[di,t] = rr_q[i0+di] . r[m0+t],  t = 0..47, m0 = j0-i0+1008
    int m0 = j0 - i0 + 1008;
#pragma unroll
    for (int t3 = 0; t3 < 3; ++t3) {
      v8f P = zero8;
#pragma unroll
      for (int c = 0; c < 2; ++c) {
        v16bf rf = load_frag_r(r32, m0 + t3 * 16, c * 16, lane);
        P = wmma_bf16(arr[c], rf, P);
      }
#pragma unroll
      for (int v = 0; v < 8; ++v)
        p_lds[wv][(v + 8 * hi) * 48 + t3 * 16 + ln] = P[v];
    }
    __syncthreads();

    // combine AC + shifted BD, mask, online softmax
    unsigned short* prw = (unsigned short*)pr_lds[wv];
#pragma unroll
    for (int v = 0; v < 8; ++v) {
      int di = v + 8 * hi;
      int ig = i0 + di;
      float s0 = (S0[v] + p_lds[wv][di * 48 + (ln - di + 15)]) * scale;
      float s1 = (S1[v] + p_lds[wv][di * 48 + (ln - di + 31)]) * scale;
      if (j0 + ln > ig + MEML) s0 = -1e30f;
      if (j0 + 16 + ln > ig + MEML) s1 = -1e30f;
      float mx = fmaxf(s0, s1);
#pragma unroll
      for (int off = 1; off < 16; off <<= 1) mx = fmaxf(mx, __shfl_xor(mx, off, 32));
      float mnew = fmaxf(mrun[v], mx);
      float alpha = __expf(mrun[v] - mnew);
      mrun[v] = mnew;
      float p0 = __expf(s0 - mnew);
      float p1 = __expf(s1 - mnew);
      float rs = p0 + p1;
#pragma unroll
      for (int off = 1; off < 16; off <<= 1) rs += __shfl_xor(rs, off, 32);
      lrun[v] = lrun[v] * alpha + rs;
      O0[v] *= alpha; O1[v] *= alpha; O2[v] *= alpha; O3[v] *= alpha;
      prw[di * 32 + ln]      = f2bf(p0);
      prw[di * 32 + 16 + ln] = f2bf(p1);
    }
    __syncthreads();

    // PV: probs (16x32) x V (32x64)
    v16bf pa = load_frag_a((const unsigned*)pr_lds[wv], 16, lane);
    v16bf vf0 = load_frag_b(v_lds + 0 * 16, 16, lane);
    v16bf vf1 = load_frag_b(v_lds + 16 * 16, 16, lane);
    v16bf vf2 = load_frag_b(v_lds + 32 * 16, 16, lane);
    v16bf vf3 = load_frag_b(v_lds + 48 * 16, 16, lane);
    O0 = wmma_bf16(pa, vf0, O0);
    O1 = wmma_bf16(pa, vf1, O1);
    O2 = wmma_bf16(pa, vf2, O2);
    O3 = wmma_bf16(pa, vf3, O3);
  }

  // epilogue: normalize and scatter to attn_vec [Q*B, NH*DH] (bf16)
#pragma unroll
  for (int v = 0; v < 8; ++v) {
    float invl = 1.0f / lrun[v];
    int row = i0 + v + 8 * hi;
    size_t base = ((size_t)row * BATCH + b) * (NHEAD * DHEAD) + n * DHEAD + ln;
    av[base + 0]  = f2bf(O0[v] * invl);
    av[base + 16] = f2bf(O1[v] * invl);
    av[base + 32] = f2bf(O2[v] * invl);
    av[base + 48] = f2bf(O3[v] * invl);
  }
}

// ---------- fused residual-add (+optional bias) + LayerNorm ------------------

__global__ __launch_bounds__(256) void add_ln_kernel(const float* __restrict__ x,
                                                     const float* __restrict__ res,
                                                     const float* __restrict__ bias,
                                                     const float* __restrict__ g,
                                                     const float* __restrict__ be,
                                                     float* __restrict__ of32,
                                                     unsigned short* __restrict__ obf) {
  __shared__ float ybuf[DMODEL];
  __shared__ float red[16];
  int row = blockIdx.x, tid = threadIdx.x;
  int lane = tid & 31, wv = tid >> 5;
  const float* xr = x + (size_t)row * DMODEL;
  const float* rr = res + (size_t)row * DMODEL;
  float s = 0.f, s2 = 0.f;
#pragma unroll
  for (int t = 0; t < 4; ++t) {
    int c = t * 256 + tid;
    float y = xr[c] + rr[c] + (bias ? bias[c] : 0.f);
    ybuf[c] = y; s += y; s2 += y * y;
  }
#pragma unroll
  for (int off = 1; off < 32; off <<= 1) { s += __shfl_xor(s, off, 32); s2 += __shfl_xor(s2, off, 32); }
  if (lane == 0) { red[wv] = s; red[8 + wv] = s2; }
  __syncthreads();
  float ts = 0.f, ts2 = 0.f;
#pragma unroll
  for (int w2 = 0; w2 < 8; ++w2) { ts += red[w2]; ts2 += red[8 + w2]; }
  float mean = ts * (1.0f / DMODEL);
  float var = ts2 * (1.0f / DMODEL) - mean * mean;
  float inv = rsqrtf(var + 1e-5f);
#pragma unroll
  for (int t = 0; t < 4; ++t) {
    int c = t * 256 + tid;
    float yv = (ybuf[c] - mean) * inv * g[c] + be[c];
    if (of32) of32[(size_t)row * DMODEL + c] = yv;
    if (obf)  obf[(size_t)row * DMODEL + c] = f2bf(yv);
  }
}

// ---------- host orchestration ----------------------------------------------

extern "C" void kernel_launch(void* const* d_in, const int* in_sizes, int n_in,
                              void* d_out, int out_size, void* d_ws, size_t ws_size,
                              hipStream_t stream) {
  (void)in_sizes; (void)n_in; (void)out_size; (void)ws_size;
  const float* w       = (const float*)d_in[0];
  const float* r       = (const float*)d_in[1];
  const float* mems    = (const float*)d_in[2];
  const float* qkv_w   = (const float*)d_in[3];
  const float* r_net_w = (const float*)d_in[4];
  const float* o_w     = (const float*)d_in[5];
  const float* r_w_bias= (const float*)d_in[6];
  const float* r_r_bias= (const float*)d_in[7];
  const float* ln1_g   = (const float*)d_in[8];
  const float* ln1_b   = (const float*)d_in[9];
  const float* ff_w1   = (const float*)d_in[10];
  const float* ff_b1   = (const float*)d_in[11];
  const float* ff_w2   = (const float*)d_in[12];
  const float* ff_b2   = (const float*)d_in[13];
  const float* ln2_g   = (const float*)d_in[14];
  const float* ln2_b   = (const float*)d_in[15];
  float* out = (float*)d_out;

  char* ws = (char*)d_ws;
  size_t off = 0;
  auto alloc = [&](size_t bytes) -> char* {
    char* p = ws + off; off = (off + bytes + 255) & ~(size_t)255; return p;
  };

  const size_t ROWS = (size_t)KLEN * BATCH;          // 8192
  const size_t QB   = (size_t)QLEN * BATCH;          // 4096

  unsigned short* cat_bf  = (unsigned short*)alloc(ROWS * DMODEL * 2);        // reused as ff2 f32
  unsigned short* qkvw_bf = (unsigned short*)alloc((size_t)DMODEL * 3072 * 2);
  unsigned short* rnet_bf = (unsigned short*)alloc((size_t)DMODEL * DMODEL * 2);
  unsigned short* ow_bf   = (unsigned short*)alloc((size_t)DMODEL * DMODEL * 2);
  unsigned short* ffw1_bf = (unsigned short*)alloc((size_t)DMODEL * DINNER * 2);
  unsigned short* ffw2_bf = (unsigned short*)alloc((size_t)DINNER * DMODEL * 2);
  unsigned short* r_bf    = (unsigned short*)alloc((size_t)KLEN * DMODEL * 2);
  float* wheads = (float*)alloc(ROWS * 3072 * 4);                             // reused as ffh f32
  float* rproj  = (float*)alloc((size_t)KLEN * DMODEL * 4);
  unsigned short* rwq  = (unsigned short*)alloc(QB * DMODEL * 2);
  unsigned short* rrq  = (unsigned short*)alloc(QB * DMODEL * 2);
  unsigned short* kbuf = (unsigned short*)alloc(ROWS * DMODEL * 2);           // kbuf+vbuf adjacent,
  unsigned short* vbuf = (unsigned short*)alloc(ROWS * DMODEL * 2);           // reused as ffh_bf
  unsigned short* rbuf = (unsigned short*)alloc((size_t)NHEAD * KLEN * DHEAD * 2);
  unsigned short* attn_vec = (unsigned short*)alloc(QB * DMODEL * 2);
  float* attn_out = (float*)alloc(QB * DMODEL * 4);
  float* ln1      = (float*)alloc(QB * DMODEL * 4);
  unsigned short* ln1bf = (unsigned short*)alloc(QB * DMODEL * 2);
  float* ffh = wheads;                         // 67MB <= 100MB, wheads dead after repack
  unsigned short* ffh_bf = kbuf;               // 33.5MB == kbuf+vbuf, dead after attn
  float* ff2 = (float*)cat_bf;                 // 16.78MB == cat region, dead after gemm1

  dim3 b256(256), b128(128);

  // casts (fp32 -> bf16)
  cast_bf16_kernel<<<dim3(3145728 / 256), b256, 0, stream>>>(qkv_w, qkvw_bf, 3145728);
  cast_bf16_kernel<<<dim3(1048576 / 256), b256, 0, stream>>>(r_net_w, rnet_bf, 1048576);
  cast_bf16_kernel<<<dim3(1048576 / 256), b256, 0, stream>>>(o_w, ow_bf, 1048576);
  cast_bf16_kernel<<<dim3(4194304 / 256), b256, 0, stream>>>(ff_w1, ffw1_bf, 4194304);
  cast_bf16_kernel<<<dim3(4194304 / 256), b256, 0, stream>>>(ff_w2, ffw2_bf, 4194304);
  cast_bf16_kernel<<<dim3(2097152 / 256), b256, 0, stream>>>(r, r_bf, 2097152);
  cast_bf16_kernel<<<dim3(4194304 / 256), b256, 0, stream>>>(mems, cat_bf, 4194304);
  cast_bf16_kernel<<<dim3(4194304 / 256), b256, 0, stream>>>(w, cat_bf + 4194304, 4194304);

  // QKV projection: [8192,1024] @ [1024,3072]
  gemm_bf16_kernel<<<dim3(3072 / 64, 8192 / 128), b128, 0, stream>>>(cat_bf, qkvw_bf, wheads, 8192, 3072, 1024);
  // r projection: [2048,1024] @ [1024,1024]
  gemm_bf16_kernel<<<dim3(1024 / 64, 2048 / 128), b128, 0, stream>>>(r_bf, rnet_bf, rproj, 2048, 1024, 1024);

  repack_qkv_kernel<<<dim3((unsigned)(ROWS * 3072 / 256)), b256, 0, stream>>>(
      wheads, r_w_bias, r_r_bias, rwq, rrq, kbuf, vbuf);
  repack_r_kernel<<<dim3((unsigned)(KLEN * DMODEL / 256)), b256, 0, stream>>>(rproj, rbuf);

  // flash attention (AC + shifted BD + PV all on WMMA)
  attn_kernel<<<dim3(BATCH * NHEAD, QLEN / 64), b128, 0, stream>>>(rwq, rrq, kbuf, vbuf, rbuf, attn_vec);

  // output projection: [4096,1024] @ [1024,1024]
  gemm_bf16_kernel<<<dim3(1024 / 64, 4096 / 128), b128, 0, stream>>>(attn_vec, ow_bf, attn_out, 4096, 1024, 1024);

  // LN1: ln1 = LayerNorm(w + attn_out)
  add_ln_kernel<<<dim3((unsigned)QB), b256, 0, stream>>>(attn_out, w, nullptr, ln1_g, ln1_b, ln1, ln1bf);

  // FFN
  gemm_bf16_kernel<<<dim3(4096 / 64, 4096 / 128), b128, 0, stream>>>(ln1bf, ffw1_bf, ffh, 4096, 4096, 1024);
  bias_relu_cast_kernel<<<dim3((unsigned)(QB * DINNER / 256)), b256, 0, stream>>>(ffh, ff_b1, ffh_bf, DINNER);
  gemm_bf16_kernel<<<dim3(1024 / 64, 4096 / 128), b128, 0, stream>>>(ffh_bf, ffw2_bf, ff2, 4096, 1024, 4096);

  // LN2: out = LayerNorm(ln1 + ff2 + ff_b2)
  add_ln_kernel<<<dim3((unsigned)QB), b256, 0, stream>>>(ff2, ln1, ff_b2, ln2_g, ln2_b, out, nullptr);
}
